// AgentSelfAttention_52398601011338
// MI455X (gfx1250) — matrix operand order repaired and checked by reading, not compiled, not run on hardware
//
#include <hip/hip_runtime.h>

typedef _Float16 half_t;
typedef __attribute__((ext_vector_type(16))) _Float16 v16h;
typedef __attribute__((ext_vector_type(8)))  float    v8f;

static constexpr int Bc  = 4;
static constexpr int Nn  = 8192;
static constexpr int DIM = 1024;
static constexpr int Hh  = 16;
static constexpr int DH  = 64;
static constexpr int Mm  = 128;
static constexpr int BN  = Bc * Nn;       // 32768
static constexpr int K3  = 3 * Hh * DH;   // 3072

__device__ __forceinline__ v8f wmma_f16(v16h a, v16h b, v8f c) {
  return __builtin_amdgcn_wmma_f32_16x16x32_f16(false, a, false, b, (short)0, c,
                                                false, false);
}

// A fragment (16x32 f16) for v_wmma_f32_16x16x32_f16.
// Source: row-major [16 rows][>=32 K], K contiguous. Per ISA layout:
//   lanes 0-15: row=lane,    K = {0..7, 16..23}
//   lanes16-31: row=lane-16, K = {8..15, 24..31}
// packed 2 f16 per VGPR with even K in [15:0] -> aligned dword loads.
__device__ __forceinline__ v16h frag_a_rm(const half_t* base, int rstride, int k0,
                                          int lane) {
  union { v16h v; unsigned int u[8]; } f;
  const half_t* p = base + (size_t)(lane & 15) * rstride + k0 + ((lane & 16) ? 8 : 0);
#pragma unroll
  for (int t = 0; t < 4; ++t) f.u[t] = *(const unsigned int*)(p + 2 * t);
#pragma unroll
  for (int t = 0; t < 4; ++t) f.u[4 + t] = *(const unsigned int*)(p + 16 + 2 * t);
  return f.v;
}

// B fragment (32x16 f16). Source stored K-fast: [16 cols][>=32 K].
//   lanes 0-15: col=lane,    K = {0..7, 16..23} ; lanes 16-31: col=lane-16, K = {8..15, 24..31}
__device__ __forceinline__ v16h frag_b_kf(const half_t* base, int cstride, int k0,
                                          int lane) {
  union { v16h v; unsigned int u[8]; } f;
  const half_t* p = base + (size_t)(lane & 15) * cstride + k0 + ((lane & 16) ? 8 : 0);
#pragma unroll
  for (int t = 0; t < 4; ++t) f.u[t] = *(const unsigned int*)(p + 2 * t);
#pragma unroll
  for (int t = 0; t < 4; ++t) f.u[4 + t] = *(const unsigned int*)(p + 16 + 2 * t);
  return f.v;
}

// ---------------------------------------------------------------- casts
__global__ void cast_kernel(const float* __restrict__ in, half_t* __restrict__ out,
                            size_t n, float scale) {
  size_t i = (size_t)blockIdx.x * blockDim.x + threadIdx.x;
  size_t st = (size_t)gridDim.x * blockDim.x;
  for (; i < n; i += st) out[i] = (half_t)(in[i] * scale);
}

// ---------------------------------------------------------------- gates
// gates[b,n,h] = sigmoid(x[b,n,:] . W_gate[:,h] + b_gate[h])
__global__ __launch_bounds__(256) void gate_kernel(const float* __restrict__ x,
                                                   const float* __restrict__ Wg,
                                                   const float* __restrict__ bg,
                                                   float* __restrict__ gates) {
  __shared__ float red[256];
  int row = blockIdx.x;  // 0..BN-1
  int tid = threadIdx.x;
  int h = tid & 15, g = tid >> 4;
  const float* xr = x + (size_t)row * DIM;
  float s = 0.f;
  for (int d = g; d < DIM; d += 16) s += xr[d] * Wg[(size_t)d * Hh + h];
  red[tid] = s;
  __syncthreads();
  if (tid < 16) {
    float t = 0.f;
#pragma unroll
    for (int gg = 0; gg < 16; ++gg) t += red[tid + (gg << 4)];
    float z = t + bg[tid];
    gates[(size_t)row * Hh + tid] = 1.f / (1.f + __expf(-z));
  }
}

// ---------------------------------------------------------------- big GEMMs
// C[128x128 tile] = A[M x K] * B[K x Ncols], f16 in, f32 acc.
// MODE 0: scatter QKV -> q[b,h,n,d], k[b,h,n,d], vT[b,h,d,n] (f16)
// MODE 1: out fp32 row-major [M x Ncols]
template <int MODE>
__global__ __launch_bounds__(256) void gemm128(const half_t* __restrict__ A,
                                               const half_t* __restrict__ Bm,
                                               int K, int Ncols,
                                               half_t* __restrict__ o_q,
                                               half_t* __restrict__ o_k,
                                               half_t* __restrict__ o_vT,
                                               float* __restrict__ o_f) {
  __shared__ half_t As[128][40];  // [row][k], padded
  __shared__ half_t Bt[128][40];  // [n][k] (transposed on fill), padded
  int tid = threadIdx.x, lane = tid & 31, w = tid >> 5;
  int wrow = (w >> 1) * 32;  // 0/32/64/96
  int wcol = (w & 1) * 64;   // 0/64
  int m0 = blockIdx.y * 128;
  int n0 = blockIdx.x * 128;
  v8f acc[2][4] = {};

  for (int k0 = 0; k0 < K; k0 += 32) {
    __syncthreads();
    {  // A tile 128x32
      int row = tid >> 1, seg = (tid & 1) * 16;
      const half_t* src = A + (size_t)(m0 + row) * K + k0 + seg;
      uint4 u0 = *(const uint4*)(src);
      uint4 u1 = *(const uint4*)(src + 8);
      *(uint4*)&As[row][seg] = u0;
      *(uint4*)&As[row][seg + 8] = u1;
    }
    {  // B tile 32x128, transposed into Bt[n][k]
      int kk = tid & 31, nseg = (tid >> 5) * 16;
      const half_t* src = Bm + (size_t)(k0 + kk) * Ncols + n0 + nseg;
      uint4 u0 = *(const uint4*)(src);
      uint4 u1 = *(const uint4*)(src + 8);
      const half_t* h0 = (const half_t*)&u0;
      const half_t* h1 = (const half_t*)&u1;
#pragma unroll
      for (int e = 0; e < 8; ++e) Bt[nseg + e][kk] = h0[e];
#pragma unroll
      for (int e = 0; e < 8; ++e) Bt[nseg + 8 + e][kk] = h1[e];
    }
    if (k0 + 32 < K) {  // global_prefetch_b8 of next K slab
      __builtin_prefetch(A + (size_t)(m0 + (tid >> 1)) * K + k0 + 32, 0, 1);
      __builtin_prefetch(Bm + (size_t)(k0 + 32 + (tid & 31)) * Ncols + n0, 0, 1);
    }
    __syncthreads();
    v16h af0 = frag_a_rm(&As[wrow][0], 40, 0, lane);
    v16h af1 = frag_a_rm(&As[wrow + 16][0], 40, 0, lane);
#pragma unroll
    for (int nt = 0; nt < 4; ++nt) {
      v16h bf = frag_b_kf(&Bt[wcol + nt * 16][0], 40, 0, lane);
      acc[0][nt] = wmma_f16(af0, bf, acc[0][nt]);
      acc[1][nt] = wmma_f16(af1, bf, acc[1][nt]);
    }
  }

#pragma unroll
  for (int mt = 0; mt < 2; ++mt) {
#pragma unroll
    for (int nt = 0; nt < 4; ++nt) {
      int gcol = n0 + wcol + nt * 16 + (lane & 15);
      int rbase = m0 + wrow + mt * 16 + ((lane & 16) ? 8 : 0);
#pragma unroll
      for (int r = 0; r < 8; ++r) {
        int grow = rbase + r;
        float val = acc[mt][nt][r];
        if (MODE == 0) {
          int b = grow >> 13, n = grow & 8191;
          int q3 = gcol >> 10, rem = gcol & 1023;
          int h = rem >> 6, d = rem & 63;
          half_t hv = (half_t)val;
          size_t bh = (size_t)b * Hh + h;
          if (q3 == 0)      o_q[(bh * Nn + n) * DH + d] = hv;
          else if (q3 == 1) o_k[(bh * Nn + n) * DH + d] = hv;
          else              o_vT[(bh * DH + d) * Nn + n] = hv;
        } else {
          o_f[(size_t)grow * Ncols + gcol] = val;
        }
      }
    }
  }
}

// ---------------------------------------------------------------- QA (fused)
// per block: (b, 16 rows of N). 8 waves; wave w does heads {w, w+8}:
//   S = q[16x64] @ a[h]^T (N=128) via WMMA, softmax over 128 in registers,
//   store p(h) f16 to LDS, then talking heads x = {w, w+8} -> qaP[b,x,i,j].
__global__ __launch_bounds__(256) void qa_kernel(const half_t* __restrict__ qh,
                                                 const half_t* __restrict__ ah,
                                                 const float* __restrict__ Wqa,
                                                 half_t* __restrict__ qaP) {
  extern __shared__ half_t sp[];  // [16 h][16 i][128 j] = 64 KB
  int tid = threadIdx.x, lane = tid & 31, w = tid >> 5;
  int blk = blockIdx.x;
  int b = blk >> 9;            // N/16 = 512 tiles per batch
  int i0 = (blk & 511) * 16;

  for (int hi = 0; hi < 2; ++hi) {
    int h = w + hi * 8;
    const half_t* qbase = qh + (((size_t)b * Hh + h) * Nn + i0) * DH;
    const half_t* abase = ah + (size_t)h * Mm * DH;
    v16h af0 = frag_a_rm(qbase, DH, 0, lane);
    v16h af1 = frag_a_rm(qbase, DH, 32, lane);
    v8f acc[8];
#pragma unroll
    for (int jt = 0; jt < 8; ++jt) {
      v8f c = {};
      v16h b0 = frag_b_kf(abase + (size_t)(jt * 16) * DH, DH, 0, lane);
      v16h b1 = frag_b_kf(abase + (size_t)(jt * 16) * DH, DH, 32, lane);
      c = wmma_f16(af0, b0, c);
      c = wmma_f16(af1, b1, c);
      acc[jt] = c;
    }
    // softmax over the 128 columns of each of this wave-half's rows
#pragma unroll
    for (int r = 0; r < 8; ++r) {
      float vals[8];
      float mx = -3.0e38f;
#pragma unroll
      for (int jt = 0; jt < 8; ++jt) { vals[jt] = acc[jt][r]; mx = fmaxf(mx, vals[jt]); }
#pragma unroll
      for (int off = 1; off < 16; off <<= 1) mx = fmaxf(mx, __shfl_xor(mx, off, 32));
      float s = 0.f;
#pragma unroll
      for (int jt = 0; jt < 8; ++jt) { vals[jt] = __expf(vals[jt] - mx); s += vals[jt]; }
#pragma unroll
      for (int off = 1; off < 16; off <<= 1) s += __shfl_xor(s, off, 32);
      float inv = 1.0f / s;
      int row = r + ((lane & 16) ? 8 : 0);
#pragma unroll
      for (int jt = 0; jt < 8; ++jt)
        sp[((size_t)h * 16 + row) * 128 + jt * 16 + (lane & 15)] =
            (half_t)(vals[jt] * inv);
    }
  }
  __syncthreads();
  // talking heads: x = w and w+8
  float w0[16], w1[16];
#pragma unroll
  for (int h2 = 0; h2 < 16; ++h2) {
    w0[h2] = Wqa[w * 16 + h2];
    w1[h2] = Wqa[(w + 8) * 16 + h2];
  }
  for (int e = lane; e < 16 * 128; e += 32) {
    int i = e >> 7, j = e & 127;
    float s0 = 0.f, s1 = 0.f;
#pragma unroll
    for (int h2 = 0; h2 < 16; ++h2) {
      float pv = (float)sp[((size_t)h2 * 16 + i) * 128 + j];
      s0 += w0[h2] * pv;
      s1 += w1[h2] * pv;
    }
    qaP[(((size_t)b * Hh + w) * Nn + i0 + i) * Mm + j] = (half_t)s0;
    qaP[(((size_t)b * Hh + w + 8) * Nn + i0 + i) * Mm + j] = (half_t)s1;
  }
}

// ---------------------------------------------------------------- AK logits
// sS[b,h,m,j] = a[h] (128x64) @ k[b,h]^T via WMMA, masked, f16.
__global__ __launch_bounds__(256) void ak_gemm(const half_t* __restrict__ ah,
                                               const half_t* __restrict__ kh,
                                               const unsigned char* __restrict__ mask,
                                               half_t* __restrict__ sS) {
  int tid = threadIdx.x, lane = tid & 31, w = tid >> 5;
  int blk = blockIdx.x;  // (b*H+h)*8 + jblock
  int jblock = blk & 7, bh = blk >> 3;
  int b = bh >> 4;
  int h = bh & 15;
  int jbase = jblock * 1024 + w * 128;
  const half_t* abase = ah + (size_t)h * Mm * DH;
  const half_t* kbase = kh + (size_t)bh * Nn * DH;  // [n][d], K(=d)-fast
  const unsigned char* mrow = mask + (size_t)b * Nn;
  half_t* srow = sS + (size_t)bh * Mm * Nn;
  for (int rb = 0; rb < 8; ++rb) {
    v16h a0 = frag_a_rm(abase + (size_t)rb * 16 * DH, DH, 0, lane);
    v16h a1 = frag_a_rm(abase + (size_t)rb * 16 * DH, DH, 32, lane);
    for (int jt = 0; jt < 8; ++jt) {
      int jcol0 = jbase + jt * 16;
      v16h b0 = frag_b_kf(kbase + (size_t)jcol0 * DH, DH, 0, lane);
      v16h b1 = frag_b_kf(kbase + (size_t)jcol0 * DH, DH, 32, lane);
      v8f c = {};
      c = wmma_f16(a0, b0, c);
      c = wmma_f16(a1, b1, c);
      int j = jcol0 + (lane & 15);
      bool ok = mrow[j] != 0;
      int r0 = rb * 16 + ((lane & 16) ? 8 : 0);
#pragma unroll
      for (int r = 0; r < 8; ++r) {
        float vv = ok ? c[r] : -60000.0f;
        srow[(size_t)(r0 + r) * Nn + j] = (half_t)vv;
      }
    }
  }
}

// in-place softmax over rows of length 8192
__global__ __launch_bounds__(256) void row_softmax_8192(half_t* __restrict__ sS) {
  __shared__ float red[256];
  int tid = threadIdx.x;
  half_t* p = sS + (size_t)blockIdx.x * Nn;
  float m = -3.0e38f;
  for (int j = tid; j < Nn; j += 256) m = fmaxf(m, (float)p[j]);
  red[tid] = m;
  __syncthreads();
  for (int s2 = 128; s2 > 0; s2 >>= 1) {
    if (tid < s2) red[tid] = fmaxf(red[tid], red[tid + s2]);
    __syncthreads();
  }
  m = red[0];
  __syncthreads();
  float s = 0.f;
  for (int j = tid; j < Nn; j += 256) s += __expf((float)p[j] - m);
  red[tid] = s;
  __syncthreads();
  for (int s2 = 128; s2 > 0; s2 >>= 1) {
    if (tid < s2) red[tid] += red[tid + s2];
    __syncthreads();
  }
  float inv = 1.f / red[0];
  for (int j = tid; j < Nn; j += 256) p[j] = (half_t)(__expf((float)p[j] - m) * inv);
}

// akP[b,x,i,j] = sum_h Wak[x,h] * pS[b,h,i,j]
__global__ __launch_bounds__(256) void ak_talk(const half_t* __restrict__ pS,
                                               const float* __restrict__ Wak,
                                               half_t* __restrict__ akP) {
  __shared__ float Wl[256];
  Wl[threadIdx.x] = Wak[threadIdx.x];
  __syncthreads();
  size_t idx = (size_t)blockIdx.x * 256 + threadIdx.x;
  size_t total = (size_t)Bc * Mm * Nn;
  if (idx >= total) return;
  int j = (int)(idx & 8191);
  int i = (int)((idx >> 13) & 127);
  int b = (int)(idx >> 20);
  size_t stride = (size_t)Mm * Nn;
  size_t base = (size_t)b * Hh * stride + (size_t)i * Nn + j;
  float pv[16];
#pragma unroll
  for (int h = 0; h < 16; ++h) pv[h] = (float)pS[base + h * stride];
#pragma unroll
  for (int x = 0; x < 16; ++x) {
    float s = 0.f;
#pragma unroll
    for (int h = 0; h < 16; ++h) s += Wl[x * 16 + h] * pv[h];
    akP[base + x * stride] = (half_t)s;
  }
}

// ---------------------------------------------------------------- ag = akP @ v
// agT[b,h,d,m] (f16), K = 8192.  vT stored [b,h,d,n] -> B fragments K-fast.
__global__ __launch_bounds__(256) void ag_gemm(const half_t* __restrict__ akP,
                                               const half_t* __restrict__ vT,
                                               half_t* __restrict__ agT) {
  int tid = threadIdx.x, lane = tid & 31, w = tid >> 5;
  int bh = blockIdx.x;  // 64
  const half_t* A = akP + (size_t)bh * Mm * Nn;  // [128][8192]
  const half_t* Bv = vT + (size_t)bh * DH * Nn;  // [64][8192]
  half_t* out = agT + (size_t)bh * DH * Mm;      // [64][128]
  int r0 = w * 16;
  v8f acc[4] = {};
  for (int k0 = 0; k0 < Nn; k0 += 32) {
    v16h af = frag_a_rm(A + (size_t)r0 * Nn, Nn, k0, lane);
#pragma unroll
    for (int ct = 0; ct < 4; ++ct) {
      v16h bf = frag_b_kf(Bv + (size_t)(ct * 16) * Nn, Nn, k0, lane);
      acc[ct] = wmma_f16(af, bf, acc[ct]);
    }
  }
#pragma unroll
  for (int ct = 0; ct < 4; ++ct) {
    int d = ct * 16 + (lane & 15);
    int mb = r0 + ((lane & 16) ? 8 : 0);
#pragma unroll
    for (int r = 0; r < 8; ++r)
      out[(size_t)d * Mm + mb + r] = (half_t)acc[ct][r];
  }
}

// ---------------------------------------------------------------- out = qaP@ag
// outpre[b,i,h*64+d] = mask[b,i] * gates[b,i,h] * (qaP[b,h,i,:] . ag[b,h,:,d])
__global__ __launch_bounds__(256) void out_gemm(const half_t* __restrict__ qaP,
                                                const half_t* __restrict__ agT,
                                                const float* __restrict__ gates,
                                                const unsigned char* __restrict__ mask,
                                                half_t* __restrict__ outpre) {
  int tid = threadIdx.x, lane = tid & 31, w = tid >> 5;
  int blk = blockIdx.x;  // (b*H+h)*64 + itile
  int itile = blk & 63, bh = blk >> 6;
  int b = bh >> 4;
  int h = bh & 15;
  const half_t* A = qaP + (size_t)bh * Nn * Mm;   // [8192][128]
  const half_t* Bm = agT + (size_t)bh * DH * Mm;  // [64][128], K-fast
  int i0 = itile * 128 + w * 16;
  v8f acc[4] = {};
  v16h af[4];
#pragma unroll
  for (int ks = 0; ks < 4; ++ks)
    af[ks] = frag_a_rm(A + (size_t)i0 * Mm, Mm, ks * 32, lane);
#pragma unroll
  for (int ct = 0; ct < 4; ++ct) {
#pragma unroll
    for (int ks = 0; ks < 4; ++ks) {
      v16h bf = frag_b_kf(Bm + (size_t)(ct * 16) * Mm, Mm, ks * 32, lane);
      acc[ct] = wmma_f16(af[ks], bf, acc[ct]);
    }
  }
#pragma unroll
  for (int ct = 0; ct < 4; ++ct) {
    int d = ct * 16 + (lane & 15);
    int ib = i0 + ((lane & 16) ? 8 : 0);
#pragma unroll
    for (int r = 0; r < 8; ++r) {
      int i = ib + r;
      float g = gates[((size_t)b * Nn + i) * Hh + h];
      float mz = mask[(size_t)b * Nn + i] ? 1.f : 0.f;
      outpre[((size_t)b * Nn + i) * DIM + h * 64 + d] = (half_t)(acc[ct][r] * g * mz);
    }
  }
}

// ---------------------------------------------------------------- launch
extern "C" void kernel_launch(void* const* d_in, const int* in_sizes, int n_in,
                              void* d_out, int out_size, void* d_ws, size_t ws_size,
                              hipStream_t stream) {
  const float* x            = (const float*)d_in[0];
  const unsigned char* mask = (const unsigned char*)d_in[1];
  const float* W_qkv        = (const float*)d_in[2];
  const float* agent_tokens = (const float*)d_in[3];
  const float* W_qa         = (const float*)d_in[4];
  const float* W_ak         = (const float*)d_in[5];
  const float* W_gate       = (const float*)d_in[6];
  const float* b_gate       = (const float*)d_in[7];
  const float* W_out        = (const float*)d_in[8];

  char* ws = (char*)d_ws;
  size_t off = 0;
  auto alloc = [&](size_t bytes) -> char* {
    char* p = ws + off;
    off += (bytes + 255) & ~(size_t)255;
    return p;
  };
  half_t* xh    = (half_t*)alloc((size_t)BN * DIM * 2);
  half_t* wqkvh = (half_t*)alloc((size_t)DIM * K3 * 2);
  half_t* wouth = (half_t*)alloc((size_t)DIM * DIM * 2);
  half_t* ah    = (half_t*)alloc((size_t)Hh * Mm * DH * 2);
  half_t* qh    = (half_t*)alloc((size_t)Bc * Hh * Nn * DH * 2);
  half_t* kh    = (half_t*)alloc((size_t)Bc * Hh * Nn * DH * 2);
  half_t* vTh   = (half_t*)alloc((size_t)Bc * Hh * DH * Nn * 2);
  half_t* qaP   = (half_t*)alloc((size_t)Bc * Hh * Nn * Mm * 2);
  half_t* sS    = (half_t*)alloc((size_t)Bc * Hh * Mm * Nn * 2);
  half_t* akP   = (half_t*)alloc((size_t)Bc * Hh * Mm * Nn * 2);
  half_t* agT   = (half_t*)alloc((size_t)Bc * Hh * DH * Mm * 2);
  float*  gates = (float*)alloc((size_t)BN * Hh * 4);
  half_t* outpre= (half_t*)alloc((size_t)BN * DIM * 2);

  cast_kernel<<<4096, 256, 0, stream>>>(x, xh, (size_t)BN * DIM, 1.0f);
  cast_kernel<<<1024, 256, 0, stream>>>(W_qkv, wqkvh, (size_t)DIM * K3, 1.0f);
  cast_kernel<<<1024, 256, 0, stream>>>(W_out, wouth, (size_t)DIM * DIM, 1.0f);
  cast_kernel<<<64, 256, 0, stream>>>(agent_tokens, ah, (size_t)Hh * Mm * DH, 0.125f);

  gate_kernel<<<BN, 256, 0, stream>>>(x, W_gate, b_gate, gates);

  gemm128<0><<<dim3(K3 / 128, BN / 128), 256, 0, stream>>>(
      xh, wqkvh, DIM, K3, qh, kh, vTh, nullptr);

  qa_kernel<<<Bc * (Nn / 16), 256, 65536, stream>>>(qh, ah, W_qa, qaP);

  ak_gemm<<<Bc * Hh * 8, 256, 0, stream>>>(ah, kh, mask, sS);
  row_softmax_8192<<<Bc * Hh * Mm, 256, 0, stream>>>(sS);
  ak_talk<<<(int)(((size_t)Bc * Mm * Nn + 255) / 256), 256, 0, stream>>>(sS, W_ak, akP);

  ag_gemm<<<Bc * Hh, 256, 0, stream>>>(akP, vTh, agT);
  out_gemm<<<Bc * Hh * (Nn / 128), 256, 0, stream>>>(qaP, agT, gates, mask, outpre);

  gemm128<1><<<dim3(DIM / 128, BN / 128), 256, 0, stream>>>(
      outpre, wouth, DIM, DIM, nullptr, nullptr, nullptr, (float*)d_out);
}
